// IntrospectiveAttention_21234318311748
// MI455X (gfx1250) — matrix operand order repaired and checked by reading, not compiled, not run on hardware
//
#include <hip/hip_runtime.h>
#include <hip/hip_bf16.h>

typedef __bf16 bf16;
typedef __attribute__((ext_vector_type(16))) __bf16 v16bf;
typedef __attribute__((ext_vector_type(8)))  __bf16 v8bf;
typedef __attribute__((ext_vector_type(8)))  float  v8f;
typedef __attribute__((ext_vector_type(4)))  unsigned v4u;
typedef __attribute__((ext_vector_type(8)))  unsigned v8u;

#define DEV __device__ __forceinline__

static constexpr int Bb = 2, Ls = 1024, Es = 1024, Hh = 16, Dd = 64, NLay = 3;
static constexpr float EPSF  = 1e-5f;
static constexpr float QSCAL = 0.125f;   // D^-0.5 = 64^-0.5

// ---------- helpers ----------

DEV bf16 f2bf(float f) {                       // round-to-nearest-even f32 -> bf16
  unsigned u = __builtin_bit_cast(unsigned, f);
  u += 0x7FFFu + ((u >> 16) & 1u);
  unsigned short h = (unsigned short)(u >> 16);
  return __builtin_bit_cast(bf16, h);
}

// B-operand fragment (32x16 bf16, KxN): lane holds column n=lane&15,
// element e -> K = (lane>>4)*16 + e  => 16 contiguous bf16 per lane (32B).
DEV v16bf load16(const bf16* p) {
  return *reinterpret_cast<const v16bf*>(p);
}

// A-operand fragment (16x32 bf16, MxK): lane holds row m=lane&15,
// element e -> K = (lane>>4)*16 + e: 16 contiguous bf16 per lane (32B).
DEV v16bf loadA(const bf16* row, int half) {
  const v8bf lo = *reinterpret_cast<const v8bf*>(row + half * 16);
  const v8bf hi = *reinterpret_cast<const v8bf*>(row + half * 16 + 8);
  return __builtin_shufflevector(lo, hi, 0,1,2,3,4,5,6,7,8,9,10,11,12,13,14,15);
}

DEV v8f wmma_bf(v16bf a, v16bf b, v8f c) {
  return __builtin_amdgcn_wmma_f32_16x16x32_bf16(false, a, false, b, (short)0, c,
                                                 false, false);
}

DEV float redmax16(float v) {
#pragma unroll
  for (int m = 1; m < 16; m <<= 1) v = fmaxf(v, __shfl_xor(v, m, 32));
  return v;
}
DEV float redsum16(float v) {
#pragma unroll
  for (int m = 1; m < 16; m <<= 1) v += __shfl_xor(v, m, 32);
  return v;
}

// Low 32 bits of a generic LDS pointer are the wave-relative LDS byte address
// (flat aperture rule).
DEV unsigned lds_off(const void* p) {
  return (unsigned)(unsigned long long)(size_t)p;
}

// ---------- Tensor Data Mover: 2-D tile (64 rows x 64 bf16) global -> LDS ----
// D# group0: count=1 | lds_addr | global_addr[56:0] | type=2 ("image")
// D# group1: data_size=1(2B), tensor_dim0/1 = 1024, tile 64x64,
//            tensor_dim0_stride = 1024 elements (row stride of W).
// Issued by one wave; completion tracked with TENSORcnt.

DEV void tdm_load_tile(const bf16* gsrc, unsigned lds_byte_off) {
  const unsigned long long ga = (unsigned long long)(const void*)gsrc;
  v4u g0;
  g0[0] = 1u;                                        // count=1, user descriptor
  g0[1] = lds_byte_off;                              // lds_addr
  g0[2] = (unsigned)ga;                              // global_addr[31:0]
  g0[3] = (unsigned)((ga >> 32) & 0x01FFFFFFull)     // global_addr[56:32]
        | (2u << 30);                                // type = 2
  v8u g1;
  g1[0] = 1u << 16;                                  // data_size = 2 bytes
  g1[1] = (1024u & 0xFFFFu) << 16;                   // tensor_dim0 lo (atomic addr 0)
  g1[2] = (1024u >> 16) | ((1024u & 0xFFFFu) << 16); // dim0 hi | tensor_dim1 lo
  g1[3] = (1024u >> 16) | (64u << 16);               // dim1 hi | tile_dim0 = 64
  g1[4] = 64u;                                       // tile_dim1 = 64, tile_dim2 = 0
  g1[5] = 1024u;                                     // tensor_dim0_stride lo
  g1[6] = 0u;                                        // stride0 hi | stride1 lo
  g1[7] = 0u;                                        // stride1 hi
  asm volatile("tensor_load_to_lds %0, %1" :: "s"(g0), "s"(g1) : "memory");
}

// ---------- f32 -> bf16 conversion ----------

__global__ __launch_bounds__(256)
void k_cvt(const float* __restrict__ s, bf16* __restrict__ d, int n) {
  for (int i = blockIdx.x * 256 + threadIdx.x; i < n; i += gridDim.x * 256)
    d[i] = f2bf(s[i]);
}

// ---------- lambda weights: sigmoid + non-affine LN over NL=3 ----------

__global__ void k_lambda(const float* __restrict__ lam, float* __restrict__ lw) {
  float s[NLay]; float mean = 0.f;
#pragma unroll
  for (int i = 0; i < NLay; ++i) { s[i] = 1.f / (1.f + __expf(-lam[i])); mean += s[i]; }
  mean *= (1.f / NLay);
  float var = 0.f;
#pragma unroll
  for (int i = 0; i < NLay; ++i) var += (s[i] - mean) * (s[i] - mean);
  var *= (1.f / NLay);
  const float r = rsqrtf(var + EPSF);
#pragma unroll
  for (int i = 0; i < NLay; ++i) lw[i] = (s[i] - mean) * r;
}

// ---------- GEMM  Y = X(2048x1024) @ W(1024x1024)^T ----------
// B tiles (64 N-rows x 64 K) staged in LDS by the TDM, double-buffered;
// 8 waves share each tile (8x traffic reduction vs direct loads).
// MODE 0: RoPE + q-scale -> bf16 (B,L,H,D)   (query)
// MODE 1: RoPE           -> bf16 (B,L,H,D)   (key)
// MODE 2: transpose      -> bf16 (B,H,D,L)   (value, key-contiguous)
// MODE 3: plain f32      -> (B,L,E)          (output projection)

template <int MODE>
__global__ __launch_bounds__(256)
void k_gemm(const bf16* __restrict__ X, const bf16* __restrict__ W,
            float* __restrict__ Yf, bf16* __restrict__ Yb,
            const float* __restrict__ cosp, const float* __restrict__ sinp) {
  __shared__ bf16 Bs[2][64][64];               // 16 KB double buffer
  const int lane = threadIdx.x & 31;
  const int wave = threadIdx.x >> 5;
  const int ln   = lane & 15;
  const int half = lane >> 4;
  const int m0   = blockIdx.x * 128 + wave * 16;
  const int n0   = blockIdx.y * 64;

  const v8f z8 = {0.f, 0.f, 0.f, 0.f, 0.f, 0.f, 0.f, 0.f};
  v8f acc[4] = {z8, z8, z8, z8};

  const bf16* xrow  = X + (size_t)(m0 + ln) * Es;
  const bf16* wtile = W + (size_t)n0 * Es;     // 64 consecutive rows of W

  if (wave == 0) {                             // prime both buffers
    tdm_load_tile(wtile + 0,  lds_off(&Bs[0][0][0]));
    tdm_load_tile(wtile + 64, lds_off(&Bs[1][0][0]));
  }

  constexpr int NIT = Es / 64;                 // 16 K-iterations
  for (int it = 0; it < NIT; ++it) {
    const int p = it & 1;
    if (wave == 0) {                           // oldest TDM op = tile `it`
      if (it < NIT - 2) __builtin_amdgcn_s_wait_tensorcnt(1);
      else              __builtin_amdgcn_s_wait_tensorcnt(0);
    }
    __syncthreads();                           // tile `it` visible to all waves

    const int kk = it * 64;
    const v16bf a0 = loadA(xrow + kk, half);
    const v16bf a1 = loadA(xrow + kk + 32, half);
#pragma unroll
    for (int nt = 0; nt < 4; ++nt) {
      const bf16* brow = &Bs[p][nt * 16 + ln][half * 16];
      acc[nt] = wmma_bf(a0, load16(brow), acc[nt]);
      acc[nt] = wmma_bf(a1, load16(brow + 32), acc[nt]);
    }
    __syncthreads();                           // all waves done reading Bs[p]
    if (wave == 0 && it + 2 < NIT)
      tdm_load_tile(wtile + (it + 2) * 64, lds_off(&Bs[p][0][0]));
  }

#pragma unroll
  for (int j = 0; j < 8; ++j) {
    const int r = half * 8 + j;          // C-layout: lanes 0-15 rows 0-7, 16-31 rows 8-15
    const int m = m0 + r;
    if (MODE == 0 || MODE == 1) {
      const int l = m & (Ls - 1);
#pragma unroll
      for (int nt = 0; nt < 4; ++nt) {
        const int d   = nt * 16 + ln;    // dim within head (n0 is head-aligned)
        const float v = acc[nt][j];
        const float rot = (nt < 2) ? -acc[nt + 2][j] : acc[nt - 2][j];
        float o = v * cosp[l * Dd + d] + rot * sinp[l * Dd + d];
        if (MODE == 0) o *= QSCAL;
        Yb[(size_t)m * Es + n0 + d] = f2bf(o);
      }
    } else if (MODE == 2) {
      const int b = m >> 10, l = m & (Ls - 1);
      const int h = n0 >> 6;
#pragma unroll
      for (int nt = 0; nt < 4; ++nt) {
        const int d = nt * 16 + ln;
        Yb[((size_t)((b * Hh + h) * Dd + d)) * Ls + l] = f2bf(acc[nt][j]);
      }
    } else {
#pragma unroll
      for (int nt = 0; nt < 4; ++nt)
        Yf[(size_t)m * Es + n0 + nt * 16 + ln] = acc[nt][j];
    }
  }
}

// ---------- flash causal attention (per layer) ----------
// grid.x = B*H, grid.y = L/128; one wave = 16 query rows; 32-key blocks.

__global__ __launch_bounds__(256)
void k_attn(const bf16* __restrict__ Q, const bf16* __restrict__ K,
            const bf16* __restrict__ Vt, float* __restrict__ A) {
  __shared__ bf16 plds[8][16][32];       // per-wave P staging (C-layout -> A-layout)
  const int lane = threadIdx.x & 31;
  const int wave = threadIdx.x >> 5;
  const int ln   = lane & 15;
  const int half = lane >> 4;
  const int b    = blockIdx.x >> 4;
  const int h    = blockIdx.x & 15;
  const int q0   = blockIdx.y * 128 + wave * 16;

  const bf16* qrow = Q + (size_t)(b * Ls + q0 + ln) * Es + h * Dd;
  const v16bf aq0 = loadA(qrow, half);        // feature dims 0..31
  const v16bf aq1 = loadA(qrow + 32, half);   // feature dims 32..63

  const v8f z8 = {0.f, 0.f, 0.f, 0.f, 0.f, 0.f, 0.f, 0.f};
  v8f o[4] = {z8, z8, z8, z8};
  float mrow[8], lrow[8];
#pragma unroll
  for (int j = 0; j < 8; ++j) { mrow[j] = -__builtin_inff(); lrow[j] = 0.f; }

  const int nb = (q0 + 47) >> 5;              // causal: keys up to q0+15
  for (int blk = 0; blk < nb; ++blk) {
    const int kb0 = blk << 5;
    v8f s[2] = {z8, z8};
    const bf16* kbase = K + (size_t)(b * Ls + kb0) * Es + h * Dd + half * 16;
#pragma unroll
    for (int nt = 0; nt < 2; ++nt) {          // 16 keys per score tile
      const bf16* kr = kbase + (size_t)(nt * 16 + ln) * Es;
      s[nt] = wmma_bf(aq0, load16(kr), s[nt]);
      s[nt] = wmma_bf(aq1, load16(kr + 32), s[nt]);
    }
    const bool needmask = (kb0 + 31) > q0;
#pragma unroll
    for (int j = 0; j < 8; ++j) {
      const int r  = half * 8 + j;
      const int qg = q0 + r;
      float x0 = s[0][j], x1 = s[1][j];
      if (needmask) {
        if (kb0 + ln > qg)      x0 = -__builtin_inff();
        if (kb0 + 16 + ln > qg) x1 = -__builtin_inff();
      }
      const float mx  = redmax16(fmaxf(x0, x1));
      const float mn  = fmaxf(mrow[j], mx);
      const float scl = __expf(mrow[j] - mn);
      const float p0  = __expf(x0 - mn);
      const float p1  = __expf(x1 - mn);
      lrow[j] = lrow[j] * scl + redsum16(p0 + p1);
      mrow[j] = mn;
      o[0][j] *= scl; o[1][j] *= scl; o[2][j] *= scl; o[3][j] *= scl;
      plds[wave][r][ln]      = f2bf(p0);
      plds[wave][r][16 + ln] = f2bf(p1);
    }
    asm volatile("s_wait_dscnt 0" ::: "memory");   // wave-local LDS RAW fence
    const v16bf ap = loadA(&plds[wave][ln][0], half);   // P as A-operand (16x32)
    const bf16* vb = Vt + ((size_t)((b * Hh + h) * Dd)) * Ls + kb0 + half * 16;
#pragma unroll
    for (int nt = 0; nt < 4; ++nt)                 // O += P @ V  (dims 0..63)
      o[nt] = wmma_bf(ap, load16(vb + (size_t)(nt * 16 + ln) * Ls), o[nt]);
  }

  float* arow = A + (size_t)(b * Ls + q0) * Es + h * Dd;
#pragma unroll
  for (int j = 0; j < 8; ++j) {
    const int r = half * 8 + j;
    const float inv = 1.0f / lrow[j];
#pragma unroll
    for (int nt = 0; nt < 4; ++nt)
      arow[(size_t)r * Es + nt * 16 + ln] = o[nt][j] * inv;
  }
}

// ---------- RMSNorm(a)*g*lw accumulate into combined ----------

__global__ __launch_bounds__(256)
void k_rms_accum(const float* __restrict__ a, const float* __restrict__ gln,
                 const float* __restrict__ lw, int layer,
                 float* __restrict__ comb, int init) {
  __shared__ float part[8];
  const int row = blockIdx.x, t = threadIdx.x;
  const float* ar = a + (size_t)row * Es;
  float v[4]; float ss = 0.f;
#pragma unroll
  for (int k = 0; k < 4; ++k) { v[k] = ar[t + k * 256]; ss += v[k] * v[k]; }
#pragma unroll
  for (int m = 1; m < 32; m <<= 1) ss += __shfl_xor(ss, m, 32);
  if ((t & 31) == 0) part[t >> 5] = ss;
  __syncthreads();
  if (t == 0) { float tt = 0.f; for (int i = 0; i < 8; ++i) tt += part[i]; part[0] = tt; }
  __syncthreads();
  const float sc = rsqrtf(part[0] * (1.f / Es) + EPSF) * lw[layer];
  float* cr = comb + (size_t)row * Es;
  const float* gr = gln + (size_t)layer * Es;
#pragma unroll
  for (int k = 0; k < 4; ++k) {
    const int c = t + k * 256;
    const float o = v[k] * sc * gr[c];
    if (init) cr[c] = o; else cr[c] += o;
  }
}

// ---------- final: rmsnorm(combined + alpha*x, g_final) -> bf16 ----------

__global__ __launch_bounds__(256)
void k_final(const float* __restrict__ comb, const float* __restrict__ x,
             const float* __restrict__ alpha, const float* __restrict__ g,
             bf16* __restrict__ yb) {
  __shared__ float part[8];
  const int row = blockIdx.x, t = threadIdx.x;
  const float al = alpha[0];
  const float* cr = comb + (size_t)row * Es;
  const float* xr = x + (size_t)row * Es;
  float v[4]; float ss = 0.f;
#pragma unroll
  for (int k = 0; k < 4; ++k) {
    v[k] = cr[t + k * 256] + al * xr[t + k * 256];
    ss += v[k] * v[k];
  }
#pragma unroll
  for (int m = 1; m < 32; m <<= 1) ss += __shfl_xor(ss, m, 32);
  if ((t & 31) == 0) part[t >> 5] = ss;
  __syncthreads();
  if (t == 0) { float tt = 0.f; for (int i = 0; i < 8; ++i) tt += part[i]; part[0] = tt; }
  __syncthreads();
  const float sc = rsqrtf(part[0] * (1.f / Es) + EPSF);
#pragma unroll
  for (int k = 0; k < 4; ++k) {
    const int c = t + k * 256;
    yb[(size_t)row * Es + c] = f2bf(v[k] * sc * g[c]);
  }
}

// ---------- launch ----------

extern "C" void kernel_launch(void* const* d_in, const int* in_sizes, int n_in,
                              void* d_out, int out_size, void* d_ws, size_t ws_size,
                              hipStream_t stream) {
  (void)in_sizes; (void)n_in; (void)out_size; (void)ws_size;
  const float* x    = (const float*)d_in[0];
  const float* cosp = (const float*)d_in[1];
  const float* sinp = (const float*)d_in[2];
  // d_in[3] attn_mask: folded analytically (causal; concatenated cache fully masked)
  const float* Wq   = (const float*)d_in[4];
  const float* Wk   = (const float*)d_in[5];
  const float* Wv   = (const float*)d_in[6];
  const float* gln  = (const float*)d_in[7];
  const float* lam  = (const float*)d_in[8];
  const float* Wo   = (const float*)d_in[9];
  const float* gfin = (const float*)d_in[10];
  const float* alp  = (const float*)d_in[11];

  char* w = (char*)d_ws;
  const size_t MB = (size_t)1 << 20;
  bf16*  xb   = (bf16*) (w + 0 * MB);    // 4 MB
  bf16*  wqb  = (bf16*) (w + 4 * MB);    // 6 MB
  bf16*  wkb  = (bf16*) (w + 10 * MB);   // 6 MB
  bf16*  wvb  = (bf16*) (w + 16 * MB);   // 6 MB
  bf16*  wob  = (bf16*) (w + 22 * MB);   // 2 MB
  bf16*  qb   = (bf16*) (w + 24 * MB);   // 4 MB
  bf16*  kbuf = (bf16*) (w + 28 * MB);   // 4 MB
  bf16*  vbt  = (bf16*) (w + 32 * MB);   // 4 MB  (B,H,D,L)
  float* ab   = (float*)(w + 36 * MB);   // 8 MB
  float* comb = (float*)(w + 44 * MB);   // 8 MB
  bf16*  yb   = (bf16*) (w + 52 * MB);   // 4 MB
  float* lw   = (float*)(w + 56 * MB);   // tiny

  k_cvt<<<dim3(2048), 256, 0, stream>>>(x,  xb,  Bb * Ls * Es);
  k_cvt<<<dim3(3072), 256, 0, stream>>>(Wq, wqb, NLay * Es * Es);
  k_cvt<<<dim3(3072), 256, 0, stream>>>(Wk, wkb, NLay * Es * Es);
  k_cvt<<<dim3(3072), 256, 0, stream>>>(Wv, wvb, NLay * Es * Es);
  k_cvt<<<dim3(1024), 256, 0, stream>>>(Wo, wob, Es * Es);
  k_lambda<<<1, 1, 0, stream>>>(lam, lw);

  const dim3 gg(16, 16), bt(256);
  for (int i = 0; i < NLay; ++i) {
    const size_t woff = (size_t)i * Es * Es;
    k_gemm<0><<<gg, bt, 0, stream>>>(xb, wqb + woff, nullptr, qb,   cosp, sinp);
    k_gemm<1><<<gg, bt, 0, stream>>>(xb, wkb + woff, nullptr, kbuf, cosp, sinp);
    k_gemm<2><<<gg, bt, 0, stream>>>(xb, wvb + woff, nullptr, vbt,  cosp, sinp);
    k_attn<<<dim3(32, 8), bt, 0, stream>>>(qb, kbuf, vbt, ab);
    k_rms_accum<<<dim3(2048), bt, 0, stream>>>(ab, gln, lw, i, comb, i == 0 ? 1 : 0);
  }
  k_final<<<dim3(2048), bt, 0, stream>>>(comb, x, alp, gfin, yb);
  k_gemm<3><<<gg, bt, 0, stream>>>(yb, wob, (float*)d_out, nullptr, cosp, sinp);
}
